// TemporalSelfAttention_38448547234009
// MI455X (gfx1250) — matrix operand order repaired and checked by reading, not compiled, not run on hardware
//
#include <hip/hip_runtime.h>
#include <hip/hip_bf16.h>

typedef __attribute__((ext_vector_type(8)))  __bf16 v8bf;
typedef __attribute__((ext_vector_type(16))) __bf16 v16bf;
typedef __attribute__((ext_vector_type(8)))  float  v8f;

#define LQ 1024
#define NSTREAM 8
#define EMB 512
#define NH 8
#define HD 64
#define ROWS (2*LQ*NSTREAM)   // 16384 flattened (b,l,n) rows

__device__ __forceinline__ v16bf cat8(v8bf a, v8bf b) {
    return __builtin_shufflevector(a, b, 0,1,2,3,4,5,6,7,8,9,10,11,12,13,14,15);
}
__device__ __forceinline__ v8bf ld8(const __bf16* p) {
    return *reinterpret_cast<const v8bf*>(p);
}
__device__ __forceinline__ v8f wmma_bf16(v16bf a, v16bf b, v8f c) {
    return __builtin_amdgcn_wmma_f32_16x16x32_bf16(false, a, false, b, (short)0, c, false, false);
}

// ---------- 1) fp32 -> bf16 convert (x) ----------
__global__ __launch_bounds__(256) void cvt_f32_bf16(const float* __restrict__ in,
                                                    __bf16* __restrict__ out, int n4) {
    int i = (blockIdx.x * 256 + threadIdx.x) * 4;
    if (i < n4) {
        float4 v = *reinterpret_cast<const float4*>(in + i);
        out[i+0] = (__bf16)v.x; out[i+1] = (__bf16)v.y;
        out[i+2] = (__bf16)v.z; out[i+3] = (__bf16)v.w;
    }
}

// ---------- 2) transpose + convert weights: w (E x F) fp32 -> wT (F x E) bf16 ----------
__global__ __launch_bounds__(256) void transpose_cvt(const float* __restrict__ w,
                                                     __bf16* __restrict__ wT, int E, int F) {
    int o = blockIdx.x * 256 + threadIdx.x;
    if (o < E * F) {
        int f = o / E, e = o - f * E;
        wT[o] = (__bf16)w[(size_t)e * F + f];
    }
}

// ---------- 3/6) GEMM: C(MxNt) = A(MxK=512) * Bt(NtxK)^T + bias ----------
template<bool OUTBF>
__global__ __launch_bounds__(256) void gemm_kernel(const __bf16* __restrict__ A,
                                                   const __bf16* __restrict__ Bt,
                                                   const float* __restrict__ bias,
                                                   __bf16* __restrict__ outb,
                                                   float* __restrict__ outf,
                                                   int M, int Nt) {
    const int K = EMB;
    int w = blockIdx.x * 8 + (threadIdx.x >> 5);
    int tilesN = Nt >> 5;
    int tm = w / tilesN, tn = w - tm * tilesN;
    if (tm >= (M >> 5)) return;
    int lane = threadIdx.x & 31, lo = lane & 15, hi = lane >> 4;

    v8f z = {0.f,0.f,0.f,0.f,0.f,0.f,0.f,0.f};
    v8f acc[2][2] = {{z, z}, {z, z}};

    const __bf16* ar0 = A  + (size_t)(tm * 32 + lo) * K;
    const __bf16* ar1 = ar0 + (size_t)16 * K;
    const __bf16* br0 = Bt + (size_t)(tn * 32 + lo) * K;
    const __bf16* br1 = br0 + (size_t)16 * K;

    for (int kb = 0; kb < K; kb += 32) {
        v16bf a0 = cat8(ld8(ar0 + kb + 8*hi), ld8(ar0 + kb + 16 + 8*hi));
        v16bf a1 = cat8(ld8(ar1 + kb + 8*hi), ld8(ar1 + kb + 16 + 8*hi));
        v16bf b0 = cat8(ld8(br0 + kb + 8*hi), ld8(br0 + kb + 16 + 8*hi));
        v16bf b1 = cat8(ld8(br1 + kb + 8*hi), ld8(br1 + kb + 16 + 8*hi));
        acc[0][0] = wmma_bf16(a0, b0, acc[0][0]);
        acc[0][1] = wmma_bf16(a0, b1, acc[0][1]);
        acc[1][0] = wmma_bf16(a1, b0, acc[1][0]);
        acc[1][1] = wmma_bf16(a1, b1, acc[1][1]);
    }
    #pragma unroll
    for (int c = 0; c < 2; ++c) {
        int col = tn * 32 + c * 16 + lo;
        float bv = bias[col];
        #pragma unroll
        for (int r = 0; r < 2; ++r) {
            int rowb = tm * 32 + r * 16 + 8 * hi;
            #pragma unroll
            for (int i = 0; i < 8; ++i) {
                float v = acc[r][c][i] + bv;
                size_t idx = (size_t)(rowb + i) * Nt + col;
                if (OUTBF) outb[idx] = (__bf16)v; else outf[idx] = v;
            }
        }
    }
}

// ---------- 4) RoPE on q,k (in place, q pre-scaled) + pack V transposed ----------
__global__ __launch_bounds__(256) void rope_pack(__bf16* __restrict__ qkv,
                                                 const int* __restrict__ pos_ids,
                                                 __bf16* __restrict__ vT) {
    int gid = blockIdx.x * 256 + threadIdx.x;          // ROWS * 256 threads
    int row = gid >> 8, rem = gid & 255;
    int h = rem >> 5, d = rem & 31;
    int b = row >> 13, l = (row >> 3) & (LQ - 1), n = row & (NSTREAM - 1);
    float pos = (float)pos_ids[b * LQ + l];
    float inv = __expf((float)d * (-2.0f / (float)HD) * 9.2103403719761836f); // ln(1e4)
    float ang = pos * inv;
    float c = __cosf(ang), s = __sinf(ang);
    size_t base = (size_t)row * (3 * EMB) + h * HD;
    float q0 = (float)qkv[base + d], q1 = (float)qkv[base + d + 32];
    qkv[base + d]      = (__bf16)((q0 * c - q1 * s) * 0.125f);  // fold 1/sqrt(hd)
    qkv[base + d + 32] = (__bf16)((q1 * c + q0 * s) * 0.125f);
    float k0 = (float)qkv[base + EMB + d], k1 = (float)qkv[base + EMB + d + 32];
    qkv[base + EMB + d]      = (__bf16)(k0 * c - k1 * s);
    qkv[base + EMB + d + 32] = (__bf16)(k1 * c + k0 * s);
    int bnh = (b * NSTREAM + n) * NH + h;
    vT[((size_t)bnh * HD + d)      * LQ + l] = qkv[base + 2 * EMB + d];
    vT[((size_t)bnh * HD + d + 32) * LQ + l] = qkv[base + 2 * EMB + d + 32];
}

// ---------- 5) causal flash attention: per (b,n,h), 16 query rows / wave ----------
__global__ __launch_bounds__(256) void attn_kernel(const __bf16* __restrict__ qkv,
                                                   const __bf16* __restrict__ vT,
                                                   __bf16* __restrict__ Y) {
    __shared__ __attribute__((aligned(16))) __bf16 lds[8 * 16 * 32];
    int bnh = blockIdx.x;
    int b = bnh >> 6, n = (bnh >> 3) & 7, h = bnh & 7;
    int wave = threadIdx.x >> 5, lane = threadIdx.x & 31;
    int lo = lane & 15, hi = lane >> 4;
    int m0 = blockIdx.y * 128 + wave * 16;

    // Q fragments (16x64 split into two 16x32 A-frags), rows m0..m0+15
    size_t qoff = ((size_t)((b * LQ + (m0 + lo)) * NSTREAM + n)) * (3 * EMB) + h * HD;
    v16bf a0 = cat8(ld8(qkv + qoff + 8 * hi),      ld8(qkv + qoff + 16 + 8 * hi));
    v16bf a1 = cat8(ld8(qkv + qoff + 32 + 8 * hi), ld8(qkv + qoff + 48 + 8 * hi));

    v8f z = {0.f,0.f,0.f,0.f,0.f,0.f,0.f,0.f};
    v8f o[4] = {z, z, z, z};
    float mr[8], sr[8];
    #pragma unroll
    for (int i = 0; i < 8; ++i) { mr[i] = -1e30f; sr[i] = 0.f; }

    __bf16* myLds = lds + wave * 512;
    int jmax = (m0 + 15) >> 5;
    for (int jt = 0; jt <= jmax; ++jt) {
        int kc = jt * 32;
        v8f s[2];
        #pragma unroll
        for (int ct = 0; ct < 2; ++ct) {
            int lk = kc + ct * 16 + lo;  // key row (lane = column of S-tile)
            size_t koff = ((size_t)((b * LQ + lk) * NSTREAM + n)) * (3 * EMB)
                        + EMB + h * HD + 8 * hi;
            v16bf bk0 = cat8(ld8(qkv + koff),      ld8(qkv + koff + 16));
            v16bf bk1 = cat8(ld8(qkv + koff + 32), ld8(qkv + koff + 48));
            s[ct] = wmma_bf16(a1, bk1, wmma_bf16(a0, bk0, z));
        }
        // exact causal mask in D layout: row = m0 + i + 8*hi, col = kc + ct*16 + lo
        #pragma unroll
        for (int i = 0; i < 8; ++i) {
            int rowi = m0 + i + 8 * hi;
            if (kc + lo > rowi)      s[0][i] = -1e30f;
            if (kc + 16 + lo > rowi) s[1][i] = -1e30f;
        }
        float p0[8], p1[8];
        #pragma unroll
        for (int i = 0; i < 8; ++i) {
            float mx = fmaxf(s[0][i], s[1][i]);
            mx = fmaxf(mx, __shfl_xor(mx, 1, 32));
            mx = fmaxf(mx, __shfl_xor(mx, 2, 32));
            mx = fmaxf(mx, __shfl_xor(mx, 4, 32));
            mx = fmaxf(mx, __shfl_xor(mx, 8, 32));
            float mn = fmaxf(mr[i], mx);
            float alpha = __expf(mr[i] - mn);
            float e0 = __expf(s[0][i] - mn);
            float e1 = __expf(s[1][i] - mn);
            float rs = e0 + e1;
            rs += __shfl_xor(rs, 1, 32);
            rs += __shfl_xor(rs, 2, 32);
            rs += __shfl_xor(rs, 4, 32);
            rs += __shfl_xor(rs, 8, 32);
            sr[i] = sr[i] * alpha + rs;
            mr[i] = mn;
            p0[i] = e0; p1[i] = e1;
            o[0][i] *= alpha; o[1][i] *= alpha; o[2][i] *= alpha; o[3][i] *= alpha;
        }
        // D-layout -> A-layout transpose of P via per-wave LDS tile (16x32 bf16)
        #pragma unroll
        for (int i = 0; i < 8; ++i) {
            myLds[(i + 8 * hi) * 32 + lo]      = (__bf16)p0[i];
            myLds[(i + 8 * hi) * 32 + lo + 16] = (__bf16)p1[i];
        }
        v16bf pf = cat8(*reinterpret_cast<v8bf*>(myLds + lo * 32 + 8 * hi),
                        *reinterpret_cast<v8bf*>(myLds + lo * 32 + 16 + 8 * hi));
        // O += P(16x32) x V(32x64)
        #pragma unroll
        for (int t = 0; t < 4; ++t) {
            const __bf16* vp = vT + ((size_t)(bnh * HD + t * 16 + lo)) * LQ + kc + 8 * hi;
            v16bf vf = cat8(ld8(vp), ld8(vp + 16));
            o[t] = wmma_bf16(pf, vf, o[t]);
        }
    }
    // finalize: divide by row sums, emit bf16 y (B,L,N,E) layout
    #pragma unroll
    for (int t = 0; t < 4; ++t) {
        #pragma unroll
        for (int i = 0; i < 8; ++i) {
            int rowl = m0 + i + 8 * hi;
            float val = o[t][i] / sr[i];
            Y[((size_t)((b * LQ + rowl) * NSTREAM + n)) * EMB + h * HD + t * 16 + lo] = (__bf16)val;
        }
    }
}

extern "C" void kernel_launch(void* const* d_in, const int* in_sizes, int n_in,
                              void* d_out, int out_size, void* d_ws, size_t ws_size,
                              hipStream_t stream) {
    (void)in_sizes; (void)n_in; (void)out_size; (void)ws_size;
    const float* x      = (const float*)d_in[0];
    const int*   posids = (const int*)  d_in[1];
    const float* w_attn = (const float*)d_in[2];
    const float* b_attn = (const float*)d_in[3];
    const float* w_proj = (const float*)d_in[4];
    const float* b_proj = (const float*)d_in[5];
    float* out = (float*)d_out;

    // workspace carve (all bf16)
    __bf16* ws  = (__bf16*)d_ws;
    const size_t XB = (size_t)ROWS * EMB;          // 8,388,608
    __bf16* xb   = ws;
    __bf16* wTa  = xb  + XB;                       // 1536 x 512
    __bf16* wTp  = wTa + (size_t)(3 * EMB) * EMB;  // 512 x 512
    __bf16* qkv  = wTp + (size_t)EMB * EMB;        // 16384 x 1536
    __bf16* vT   = qkv + (size_t)ROWS * 3 * EMB;   // 128*64 x 1024
    __bf16* y    = vT  + (size_t)128 * HD * LQ;    // 16384 x 512

    // 1) x -> bf16
    cvt_f32_bf16<<<(int)(XB / 4 / 256), 256, 0, stream>>>(x, xb, (int)XB);
    // 2) weights -> transposed bf16
    transpose_cvt<<<(3 * EMB * EMB + 255) / 256, 256, 0, stream>>>(w_attn, wTa, EMB, 3 * EMB);
    transpose_cvt<<<(EMB * EMB + 255) / 256, 256, 0, stream>>>(w_proj, wTp, EMB, EMB);
    // 3) QKV GEMM (16384x512)x(512x1536) + bias -> bf16
    gemm_kernel<true><<<(ROWS / 32) * (3 * EMB / 32) / 8, 256, 0, stream>>>(
        xb, wTa, b_attn, qkv, nullptr, ROWS, 3 * EMB);
    // 4) RoPE + V transpose-pack
    rope_pack<<<ROWS, 256, 0, stream>>>(qkv, posids, vT);
    // 5) causal flash attention
    attn_kernel<<<dim3(128, LQ / 128), 256, 0, stream>>>(qkv, vT, y);
    // 6) output projection -> fp32 d_out
    gemm_kernel<false><<<(ROWS / 32) * (EMB / 32) / 8, 256, 0, stream>>>(
        y, wTp, b_proj, nullptr, out, ROWS, EMB);
}
